// AttentionBlock_41223096107734
// MI455X (gfx1250) — matrix operand order repaired and checked by reading, not compiled, not run on hardware
//
#include <hip/hip_runtime.h>

#define BS   8
#define SEQ  2048
#define CH   128
#define ED   64
#define ROWS_TOTAL (BS*SEQ)          /* 16384 */
#define LNS  132                     /* LDS row stride (floats / halves) */
#define OUT_ATT_OFF ((size_t)ROWS_TOTAL*CH)   /* 2,097,152 floats */

typedef __attribute__((ext_vector_type(16))) _Float16 v16h;
typedef __attribute__((ext_vector_type(8)))  float    v8f;

union FragAB { v16h v; _Float16 h[16]; unsigned u[8]; };
union FragC  { v8f  v; float f[8]; };

// --- WMMA fragment k maps (contiguous k = 16*half + 2v, same for A and B; 16-bit 16x32 A / 32x16 B) ---
__device__ __forceinline__ int kmapA(int v, int half) {
  return half*16 + 2*v;
}
__device__ __forceinline__ int kmapB(int v, int half) {
  return half*16 + 2*v;
}
__device__ __forceinline__ unsigned pkh(float a, float b) {
  union { _Float16 h[2]; unsigned u; } x;
  x.h[0] = (_Float16)a; x.h[1] = (_Float16)b;
  return x.u;
}
__device__ __forceinline__ float lrelu(float x) { return x > 0.f ? x : 0.01f*x; }
__device__ __forceinline__ void lds_fence() {
  asm volatile("s_wait_dscnt 0" ::: "memory");
}
__device__ __forceinline__ v8f wmma16(const FragAB& a, const FragAB& b, const v8f& c) {
  return __builtin_amdgcn_wmma_f32_16x16x32_f16(false, a.v, false, b.v, (short)0, c,
                                                false, false);
}

// ============================================================================
// K1: x_upd = x + enc @ Wenc^T + benc ; LN1 ; h = xn @ W0^T ; s1,s2 = h@a1,h@a2
//     also stores h in a pre-swizzled f16 B-fragment layout (h_pack).
// 1 wave = one 16-row tile; block = 4 waves; grid = 256.
// ============================================================================
__global__ __launch_bounds__(128) void k1_pre(
    const float* __restrict__ x, const float* __restrict__ enc,
    const float* __restrict__ Wenc, const float* __restrict__ benc,
    const float* __restrict__ g1, const float* __restrict__ b1,
    const float* __restrict__ W0, const float* __restrict__ Wa,
    float* __restrict__ x_upd, unsigned* __restrict__ h_pack,
    float* __restrict__ s1, float* __restrict__ s2)
{
  __shared__ __align__(16) float    s_f32[4*16*LNS];
  __shared__ __align__(16) _Float16 s_f16[4*16*LNS];
  const int wid  = threadIdx.x >> 5;
  const int L    = threadIdx.x & 31;
  const int m    = L & 15, half = L >> 4;
  const int bi0  = (blockIdx.x*4 + wid) * 16;
  float*    xt = s_f32 + wid*16*LNS;
  _Float16* xn = s_f16 + wid*16*LNS;

  // ---- A fragments of enc (16x64 -> two K-chunks of 32) ----
  FragAB Ae[2];
  #pragma unroll
  for (int ch = 0; ch < 2; ++ch)
    #pragma unroll
    for (int v = 0; v < 8; ++v) {
      int k = ch*32 + kmapA(v, half);
      float2 e2 = *(const float2*)(enc + (size_t)(bi0+m)*ED + k);
      Ae[ch].u[v] = pkh(e2.x, e2.y);
    }

  // ---- x_upd = x + enc @ Wenc^T + benc ----
  #pragma unroll
  for (int ct = 0; ct < 8; ++ct) {
    FragC acc;
    #pragma unroll
    for (int i = 0; i < 8; ++i) acc.f[i] = 0.f;
    #pragma unroll
    for (int ch = 0; ch < 2; ++ch) {
      FragAB B;
      #pragma unroll
      for (int v = 0; v < 8; ++v) {
        int k  = ch*32 + kmapB(v, half);
        int ng = ct*16 + m;
        float2 w2 = *(const float2*)(Wenc + (size_t)ng*ED + k);
        B.u[v] = pkh(w2.x, w2.y);
      }
      acc.v = wmma16(Ae[ch], B, acc.v);
    }
    const int ng = ct*16 + m;
    const float be = benc[ng];
    #pragma unroll
    for (int v = 0; v < 8; ++v) {
      int mr = v + 8*half;
      size_t gi = (size_t)(bi0+mr)*CH + ng;
      float val = acc.f[v] + x[gi] + be;
      x_upd[gi] = val;
      xt[mr*LNS + ng] = val;
    }
  }
  lds_fence();

  // ---- LayerNorm 1 (two lanes per row, shfl-combine across half-wave) ----
  {
    const int row = m, c0 = half*64;
    float s = 0.f, sq = 0.f;
    #pragma unroll 8
    for (int t = 0; t < 64; ++t) { float vv = xt[row*LNS + c0 + t]; s += vv; sq += vv*vv; }
    s  += __shfl_xor(s, 16, 32);
    sq += __shfl_xor(sq, 16, 32);
    float mean = s * (1.f/CH);
    float var  = sq * (1.f/CH) - mean*mean;
    float rstd = rsqrtf(var + 1e-5f);
    #pragma unroll 8
    for (int t = 0; t < 64; ++t) {
      int col = c0 + t;
      float vv = (xt[row*LNS + col] - mean)*rstd*g1[col] + b1[col];
      xn[row*LNS + col] = (_Float16)vv;
    }
  }
  lds_fence();

  // ---- A fragments of x_norm (16x128 -> four K-chunks) ----
  FragAB Ax[4];
  #pragma unroll
  for (int ch = 0; ch < 4; ++ch)
    #pragma unroll
    for (int v = 0; v < 8; ++v) {
      int k = ch*32 + kmapA(v, half);   // k even, LNS even -> 4B aligned
      Ax[ch].u[v] = *(const unsigned*)&xn[m*LNS + k];
    }

  // ---- h = x_norm @ W0^T (f32 tile into xt, reused) ----
  #pragma unroll
  for (int ct = 0; ct < 8; ++ct) {
    FragC acc;
    #pragma unroll
    for (int i = 0; i < 8; ++i) acc.f[i] = 0.f;
    #pragma unroll
    for (int ch = 0; ch < 4; ++ch) {
      FragAB B;
      #pragma unroll
      for (int v = 0; v < 8; ++v) {
        int k  = ch*32 + kmapB(v, half);
        int ng = ct*16 + m;
        float2 w2 = *(const float2*)(W0 + (size_t)ng*CH + k);
        B.u[v] = pkh(w2.x, w2.y);
      }
      acc.v = wmma16(Ax[ch], B, acc.v);
    }
    #pragma unroll
    for (int v = 0; v < 8; ++v)
      xt[(v + 8*half)*LNS + ct*16 + m] = acc.f[v];
  }
  lds_fence();

  // ---- s1 = h@a1, s2 = h@a2 ----
  {
    const int row = m, c0 = half*64;
    float p1 = 0.f, p2 = 0.f;
    #pragma unroll 8
    for (int t = 0; t < 64; ++t) {
      int col = c0 + t;
      float hv = xt[row*LNS + col];
      p1 += hv * Wa[col];
      p2 += hv * Wa[CH + col];
    }
    p1 += __shfl_xor(p1, 16, 32);
    p2 += __shfl_xor(p2, 16, 32);
    if (half == 0) { s1[bi0+row] = p1; s2[bi0+row] = p2; }
  }

  // ---- pack h (f16) into WMMA-B-fragment global layout ----
  {
    const int b     = bi0 >> 11;
    const int iinb  = bi0 & (SEQ-1);
    const int chunk = iinb >> 5;           // 32-row K-chunk within batch
    const int thalf = (iinb >> 4) & 1;     // which half of the chunk this tile is
    unsigned* dst = h_pack + ((size_t)b*64 + chunk)*2048;
    #pragma unroll
    for (int t = 0; t < 32; ++t) {
      int f  = L*32 + t;
      int ct = f >> 7, rem = f & 127;
      int Lp = rem >> 3, v = rem & 7;
      unsigned val = pkh(xt[(2*v  )*LNS + ct*16 + Lp],
                         xt[(2*v+1)*LNS + ct*16 + Lp]);
      dst[(size_t)ct*256 + (size_t)(thalf*16 + Lp)*8 + v] = val;
    }
  }
}

// ============================================================================
// K2a: per-batch max of s2 (monotone lrelu => row max = lrelu(s1_i + max s2))
// ============================================================================
__global__ __launch_bounds__(256) void k2a_max(const float* __restrict__ s2,
                                               float* __restrict__ s2max)
{
  __shared__ float red[256];
  const int b = blockIdx.x;
  float mx = -3.4e38f;
  for (int t = threadIdx.x; t < SEQ; t += 256) mx = fmaxf(mx, s2[(size_t)b*SEQ + t]);
  red[threadIdx.x] = mx;
  __syncthreads();
  for (int s = 128; s > 0; s >>= 1) {
    if (threadIdx.x < s) red[threadIdx.x] = fmaxf(red[threadIdx.x], red[threadIdx.x+s]);
    __syncthreads();
  }
  if (threadIdx.x == 0) s2max[b] = red[0];
}

// ============================================================================
// K2b: softmax denominators. One wave per row; block caches s2[batch] in LDS.
// ============================================================================
__global__ __launch_bounds__(256) void k2b_z(
    const float* __restrict__ s1, const float* __restrict__ s2,
    const float* __restrict__ s2max,
    float* __restrict__ invZ, float* __restrict__ mrow)
{
  __shared__ float s2sh[SEQ];
  const int row0 = blockIdx.x * 8;
  const int b = row0 >> 11;
  for (int t = threadIdx.x; t < SEQ; t += 256) s2sh[t] = s2[(size_t)b*SEQ + t];
  __syncthreads();
  const int wid = threadIdx.x >> 5;
  const int L   = threadIdx.x & 31;
  const int row = row0 + wid;
  const float s1v = s1[row];
  const float mx  = lrelu(s1v + s2max[b]);
  float acc = 0.f;
  #pragma unroll 8
  for (int t = 0; t < 64; ++t)
    acc += __expf(lrelu(s1v + s2sh[t*32 + L]) - mx);
  #pragma unroll
  for (int off = 16; off > 0; off >>= 1) acc += __shfl_xor(acc, off, 32);
  if (L == 0) { invZ[row] = 1.0f / acc; mrow[row] = mx; }
}

// ============================================================================
// K3: fused softmax + atten write + attentioned = atten @ h (f16 WMMA).
// SPLIT-K: block = one 16-row i-tile, 4 waves each own 16 of 64 K-chunks
// (j-range of 512). atten writes partition by j; partial 16x128 f32
// accumulators are summed through LDS at the end. 1024 blocks -> 4096 waves.
// attentioned goes into out[0 : ROWS*CH] (h_next region, rewritten by K4).
// ============================================================================
__global__ __launch_bounds__(128) void k3_attn(
    const float* __restrict__ s1, const float* __restrict__ s2,
    const float* __restrict__ invZ, const float* __restrict__ mrow,
    const unsigned* __restrict__ h_pack, float* __restrict__ out)
{
  __shared__ __align__(16) float s2_sh[SEQ];           //  8 KiB
  __shared__ __align__(16) float plds[4*16*34];        //  8.5 KiB
  __shared__ __align__(16) float s_red[4*16*LNS];      // 33 KiB partial tiles
  const int wid = threadIdx.x >> 5;
  const int L   = threadIdx.x & 31;
  const int m   = L & 15, half = L >> 4;
  const int b   = blockIdx.x >> 7;                 // 128 i-tiles per batch
  const int i0  = (blockIdx.x & 127) * 16;
  const int bi0 = b*SEQ + i0;

  for (int t = threadIdx.x; t < SEQ; t += 128) s2_sh[t] = s2[(size_t)b*SEQ + t];
  __syncthreads();

  const float s1v = s1[bi0 + m];
  const float mv  = mrow[bi0 + m];
  const float izv = invZ[bi0 + m];

  FragC acc[8];
  #pragma unroll
  for (int ct = 0; ct < 8; ++ct)
    #pragma unroll
    for (int i = 0; i < 8; ++i) acc[ct].f[i] = 0.f;

  float* P = plds + wid*16*34;
  float* attn = out + OUT_ATT_OFF + ((size_t)b*SEQ + i0)*SEQ;
  const unsigned* hb = h_pack + (size_t)b*64*2048;

  const int c0 = wid*16, c1 = c0 + 16;      // this wave's K-chunk range
  for (int chunk = c0; chunk < c1; ++chunk) {
    const int j0 = chunk*32;
    // build probability A-fragment directly; row-major copy in LDS for output
    FragAB A;
    #pragma unroll
    for (int v = 0; v < 8; ++v) {
      int k = kmapA(v, half);
      float p0 = __expf(lrelu(s1v + s2_sh[j0+k])   - mv) * izv;
      float p1 = __expf(lrelu(s1v + s2_sh[j0+k+1]) - mv) * izv;
      *(float2*)&P[m*34 + k] = make_float2(p0, p1);
      A.u[v] = pkh(p0, p1);
    }
    lds_fence();
    // coalesced write of atten tile (16 x 32 f32)
    #pragma unroll
    for (int it = 0; it < 16; ++it)
      attn[(size_t)it*SEQ + j0 + L] = P[it*34 + L];

    const unsigned* hc = hb + (size_t)chunk*2048;
    if (chunk + 1 < c1)
      __builtin_prefetch((const void*)(hb + (size_t)(chunk+1)*2048 + L*16), 0, 1);
    #pragma unroll
    for (int ct = 0; ct < 8; ++ct) {
      FragAB B;   // pre-swizzled: two b128 loads per lane
      const uint4* hp = (const uint4*)(hc + ((size_t)ct*32 + L)*8);
      uint4 lo = hp[0], hi = hp[1];
      B.u[0]=lo.x; B.u[1]=lo.y; B.u[2]=lo.z; B.u[3]=lo.w;
      B.u[4]=hi.x; B.u[5]=hi.y; B.u[6]=hi.z; B.u[7]=hi.w;
      acc[ct].v = wmma16(A, B, acc[ct].v);
    }
    lds_fence();
  }

  // ---- split-K reduction through LDS, coalesced f32 store of attentioned ----
  {
    float* red = s_red + wid*16*LNS;
    #pragma unroll
    for (int ct = 0; ct < 8; ++ct)
      #pragma unroll
      for (int v = 0; v < 8; ++v)
        red[(v + 8*half)*LNS + ct*16 + m] = acc[ct].f[v];
  }
  __syncthreads();
  for (int t = threadIdx.x; t < 16*CH; t += 128) {
    int r = t >> 7, c = t & (CH-1);
    float s = s_red[           r*LNS + c] + s_red[  16*LNS + r*LNS + c]
            + s_red[2*16*LNS + r*LNS + c] + s_red[3*16*LNS + r*LNS + c];
    out[(size_t)(bi0 + r)*CH + c] = s;
  }
}

// ============================================================================
// K4: residual + LN2 + h_next = elu(attn_norm @ W1^T); overwrites out[0:R*CH].
// ============================================================================
__global__ __launch_bounds__(128) void k4_post(
    const float* __restrict__ x_upd, const float* __restrict__ g2,
    const float* __restrict__ b2, const float* __restrict__ W1,
    float* __restrict__ out)
{
  __shared__ __align__(16) float    s_f32[4*16*LNS];
  __shared__ __align__(16) _Float16 s_f16[4*16*LNS];
  const int wid = threadIdx.x >> 5;
  const int L   = threadIdx.x & 31;
  const int m   = L & 15, half = L >> 4;
  const int bi0 = (blockIdx.x*4 + wid) * 16;
  float*    xt = s_f32 + wid*16*LNS;
  _Float16* xn = s_f16 + wid*16*LNS;

  // residual (coalesced): attentioned (staged in out) + x_upd
  #pragma unroll
  for (int r = 0; r < 16; ++r)
    #pragma unroll
    for (int q = 0; q < 4; ++q) {
      int col = q*32 + L;
      size_t gi = (size_t)(bi0+r)*CH + col;
      xt[r*LNS + col] = out[gi] + x_upd[gi];
    }
  lds_fence();

  // LayerNorm 2
  {
    const int row = m, c0 = half*64;
    float s = 0.f, sq = 0.f;
    #pragma unroll 8
    for (int t = 0; t < 64; ++t) { float vv = xt[row*LNS + c0 + t]; s += vv; sq += vv*vv; }
    s  += __shfl_xor(s, 16, 32);
    sq += __shfl_xor(sq, 16, 32);
    float mean = s * (1.f/CH);
    float var  = sq * (1.f/CH) - mean*mean;
    float rstd = rsqrtf(var + 1e-5f);
    #pragma unroll 8
    for (int t = 0; t < 64; ++t) {
      int col = c0 + t;
      float vv = (xt[row*LNS + col] - mean)*rstd*g2[col] + b2[col];
      xn[row*LNS + col] = (_Float16)vv;
    }
  }
  lds_fence();

  FragAB Ax[4];
  #pragma unroll
  for (int ch = 0; ch < 4; ++ch)
    #pragma unroll
    for (int v = 0; v < 8; ++v) {
      int k = ch*32 + kmapA(v, half);
      Ax[ch].u[v] = *(const unsigned*)&xn[m*LNS + k];
    }

  #pragma unroll
  for (int ct = 0; ct < 8; ++ct) {
    FragC acc;
    #pragma unroll
    for (int i = 0; i < 8; ++i) acc.f[i] = 0.f;
    #pragma unroll
    for (int ch = 0; ch < 4; ++ch) {
      FragAB B;
      #pragma unroll
      for (int v = 0; v < 8; ++v) {
        int k  = ch*32 + kmapB(v, half);
        int ng = ct*16 + m;
        float2 w2 = *(const float2*)(W1 + (size_t)ng*CH + k);
        B.u[v] = pkh(w2.x, w2.y);
      }
      acc.v = wmma16(Ax[ch], B, acc.v);
    }
    const int ng = ct*16 + m;
    #pragma unroll
    for (int v = 0; v < 8; ++v) {
      float val = acc.f[v];
      val = val > 0.f ? val : (__expf(val) - 1.f);   // ELU
      out[(size_t)(bi0 + v + 8*half)*CH + ng] = val;
    }
  }
}

// ============================================================================
extern "C" void kernel_launch(void* const* d_in, const int* in_sizes, int n_in,
                              void* d_out, int out_size, void* d_ws, size_t ws_size,
                              hipStream_t stream) {
  (void)in_sizes; (void)n_in; (void)out_size; (void)ws_size;
  const float* x    = (const float*)d_in[0];
  const float* enc  = (const float*)d_in[1];
  const float* Wenc = (const float*)d_in[2];
  const float* benc = (const float*)d_in[3];
  const float* g1   = (const float*)d_in[4];
  const float* b1   = (const float*)d_in[5];
  const float* g2   = (const float*)d_in[6];
  const float* b2   = (const float*)d_in[7];
  const float* W0   = (const float*)d_in[8];
  const float* Wa   = (const float*)d_in[9];
  const float* W1   = (const float*)d_in[10];
  float* out = (float*)d_out;

  char* ws = (char*)d_ws;
  size_t off = 0;
  float*    x_upd  = (float*)(ws + off);    off += (size_t)ROWS_TOTAL*CH*4;   // 8 MiB
  unsigned* h_pack = (unsigned*)(ws + off); off += (size_t)ROWS_TOTAL*CH/2*4; // 4 MiB
  float*    s1v    = (float*)(ws + off);    off += (size_t)ROWS_TOTAL*4;
  float*    s2v    = (float*)(ws + off);    off += (size_t)ROWS_TOTAL*4;
  float*    s2mx   = (float*)(ws + off);    off += 256;
  float*    invZ   = (float*)(ws + off);    off += (size_t)ROWS_TOTAL*4;
  float*    mrow   = (float*)(ws + off);    off += (size_t)ROWS_TOTAL*4;

  k1_pre <<<dim3(ROWS_TOTAL/64), dim3(128), 0, stream>>>(x, enc, Wenc, benc, g1, b1,
                                                         W0, Wa, x_upd, h_pack, s1v, s2v);
  k2a_max<<<dim3(BS),            dim3(256), 0, stream>>>(s2v, s2mx);
  k2b_z  <<<dim3(ROWS_TOTAL/8),  dim3(256), 0, stream>>>(s1v, s2v, s2mx, invZ, mrow);
  k3_attn<<<dim3(ROWS_TOTAL/16), dim3(128), 0, stream>>>(s1v, s2v, invZ, mrow, h_pack, out);
  k4_post<<<dim3(ROWS_TOTAL/64), dim3(128), 0, stream>>>(x_upd, g2, b2, W1, out);
}
